// QuantBertSelfOutput_46815143526853
// MI455X (gfx1250) — matrix operand order repaired and checked by reading, not compiled, not run on hardware
//
#include <hip/hip_runtime.h>

typedef __attribute__((ext_vector_type(16))) _Float16 v16h;
typedef __attribute__((ext_vector_type(8)))  float    v8f;

#define LN_EPS 1e-12f

// ---------------------------------------------------------------------------
// Kernel 1: fp32 -> fp16 conversion of hidden_states (vectorized, 4/thread)
// ---------------------------------------------------------------------------
struct H4 { _Float16 a, b, c, d; };   // 8-byte packed store

__global__ __launch_bounds__(256) void cvt_f32_to_f16_kernel(
    const float* __restrict__ x, _Float16* __restrict__ y, long long n)
{
    long long i = ((long long)blockIdx.x * blockDim.x + threadIdx.x) * 4;
    if (i + 3 >= n) return;
    float4 v = *reinterpret_cast<const float4*>(x + i);
    H4 h;
    h.a = (_Float16)v.x; h.b = (_Float16)v.y;
    h.c = (_Float16)v.z; h.d = (_Float16)v.w;
    *reinterpret_cast<H4*>(y + i) = h;
}

// ---------------------------------------------------------------------------
// Kernel 2: per-output-channel weight quantization.
// Stores INTEGER-valued f16 (exact) and the per-row scale separately;
// the scale is applied in the GEMM epilogue.
// ---------------------------------------------------------------------------
__global__ __launch_bounds__(256) void quant_weight_kernel(
    const float* __restrict__ W, _Float16* __restrict__ Wq,
    float* __restrict__ scale, int D)
{
    const int row = blockIdx.x;
    const float* wr = W + (size_t)row * D;

    float m = 0.0f;
    for (int c = threadIdx.x; c < D; c += blockDim.x)
        m = fmaxf(m, fabsf(wr[c]));

    #pragma unroll
    for (int off = 16; off > 0; off >>= 1)
        m = fmaxf(m, __shfl_xor(m, off, 32));

    __shared__ float red[8];
    const int wave = threadIdx.x >> 5;
    if ((threadIdx.x & 31) == 0) red[wave] = m;
    __syncthreads();
    float mm = 0.0f;
    #pragma unroll
    for (int i = 0; i < 8; ++i) mm = fmaxf(mm, red[i]);

    const float s = fmaxf(mm * (1.0f / 127.0f), 1e-8f);
    if (threadIdx.x == 0) scale[row] = s;

    const float inv_s = 1.0f / s;
    _Float16* wqr = Wq + (size_t)row * D;
    for (int c = threadIdx.x; c < D; c += blockDim.x) {
        float q = rintf(wr[c] * inv_s);
        q = fminf(fmaxf(q, -128.0f), 127.0f);
        wqr[c] = (_Float16)q;           // exact in fp16
    }
}

// ---------------------------------------------------------------------------
// Kernel 3: WMMA GEMM  z = (X @ Wq^T) * scale + bias + residual
// Software-pipelined: fragments for K-step k+32 load while WMMAs consume k.
// Block: 256 threads = 8 waves. Block tile 128(M) x 64(N); wave tile 16x64.
// ---------------------------------------------------------------------------
union Frag { v16h v; float4 f[2]; };

#define LOAD_A(dst, kt) do {                                          \
    const _Float16* pa_ = xrow + (kt) + half_sel * 16;                \
    (dst).f[0] = *reinterpret_cast<const float4*>(pa_);               \
    (dst).f[1] = *reinterpret_cast<const float4*>(pa_ + 8);           \
} while (0)

#define LOAD_B(dst, t, kt) do {                                       \
    const float4* pb_ = reinterpret_cast<const float4*>(pb##t + (kt)); \
    (dst).f[0] = pb_[0]; (dst).f[1] = pb_[1];                         \
} while (0)

#define LOAD_ALL(a, b, kt) do {                                       \
    LOAD_A(a, kt);                                                    \
    LOAD_B(b[0], 0, kt); LOAD_B(b[1], 1, kt);                         \
    LOAD_B(b[2], 2, kt); LOAD_B(b[3], 3, kt);                         \
} while (0)

#define MMA_ALL(a, b) do {                                            \
    acc[0] = __builtin_amdgcn_wmma_f32_16x16x32_f16(                  \
        false, (a).v, false, (b)[0].v, (short)0, acc[0], false, false); \
    acc[1] = __builtin_amdgcn_wmma_f32_16x16x32_f16(                  \
        false, (a).v, false, (b)[1].v, (short)0, acc[1], false, false); \
    acc[2] = __builtin_amdgcn_wmma_f32_16x16x32_f16(                  \
        false, (a).v, false, (b)[2].v, (short)0, acc[2], false, false); \
    acc[3] = __builtin_amdgcn_wmma_f32_16x16x32_f16(                  \
        false, (a).v, false, (b)[3].v, (short)0, acc[3], false, false); \
} while (0)

__global__ __launch_bounds__(256) void gemm_wmma_kernel(
    const _Float16* __restrict__ X,
    const _Float16* __restrict__ Wq,
    const float* __restrict__ scale,
    const float* __restrict__ bias,
    const float* __restrict__ resid,
    float* __restrict__ out,
    int M, int N, int K)
{
    const int lane     = threadIdx.x & 31;
    const int wave     = threadIdx.x >> 5;
    const int half_sel = (lane >> 4) & 1;   // 0: lanes 0-15, 1: lanes 16-31
    const int l15      = lane & 15;

    const int m_wave = blockIdx.y * 128 + wave * 16;
    const int n_base = blockIdx.x * 64;

    v8f acc[4];
    #pragma unroll
    for (int t = 0; t < 4; ++t) { v8f z = {}; acc[t] = z; }

    // A: lanes 0-15 hold K 0-15, lanes 16-31 hold K 16-31
    const _Float16* xrow = X + (size_t)(m_wave + l15) * K;
    // B: lane l15 = column n; lanes 0-15 hold K 0-15, lanes 16-31 hold K 16-31
    const _Float16* pb0 = Wq + (size_t)(n_base +  0 + l15) * K + half_sel * 16;
    const _Float16* pb1 = Wq + (size_t)(n_base + 16 + l15) * K + half_sel * 16;
    const _Float16* pb2 = Wq + (size_t)(n_base + 32 + l15) * K + half_sel * 16;
    const _Float16* pb3 = Wq + (size_t)(n_base + 48 + l15) * K + half_sel * 16;

    Frag a0, a1, b0[4], b1[4];

    // Prologue: stage K-step 0
    LOAD_ALL(a0, b0, 0);

    // Ping-pong main loop: 2 K-steps per iteration, loads overlap WMMAs
    for (int kt = 0; kt < K - 64; kt += 64) {
        LOAD_ALL(a1, b1, kt + 32);
        MMA_ALL(a0, b0);
        LOAD_ALL(a0, b0, kt + 64);
        MMA_ALL(a1, b1);
    }
    // Epilogue: steps K-64 (in a0/b0) and K-32
    LOAD_ALL(a1, b1, K - 32);
    MMA_ALL(a0, b0);
    MMA_ALL(a1, b1);

    // C/D layout: VGPR r -> lanes 0-15: M=r, lanes 16-31: M=r+8; N=l15
    #pragma unroll
    for (int t = 0; t < 4; ++t) {
        const int   n  = n_base + t * 16 + l15;
        const float sc = scale[n];
        const float bb = bias[n];
        #pragma unroll
        for (int r = 0; r < 8; ++r) {
            const int m = m_wave + r + half_sel * 8;
            const size_t idx = (size_t)m * N + n;
            out[idx] = acc[t][r] * sc + bb + resid[idx];
        }
    }
}

// ---------------------------------------------------------------------------
// Kernel 4: in-place LayerNorm over rows of D=1024 (256 thr x 4 elems)
// ---------------------------------------------------------------------------
__global__ __launch_bounds__(256) void layernorm_kernel(
    float* __restrict__ z, const float* __restrict__ g,
    const float* __restrict__ be, int D)
{
    float* zr = z + (size_t)blockIdx.x * D;
    const int c = threadIdx.x * 4;

    float4 v = *reinterpret_cast<const float4*>(zr + c);
    float s  = v.x + v.y + v.z + v.w;
    float ss = v.x * v.x + v.y * v.y + v.z * v.z + v.w * v.w;

    #pragma unroll
    for (int off = 16; off > 0; off >>= 1) {
        s  += __shfl_xor(s,  off, 32);
        ss += __shfl_xor(ss, off, 32);
    }
    __shared__ float rs[8], rss[8];
    const int wave = threadIdx.x >> 5;
    if ((threadIdx.x & 31) == 0) { rs[wave] = s; rss[wave] = ss; }
    __syncthreads();
    float ts = 0.0f, tss = 0.0f;
    #pragma unroll
    for (int i = 0; i < 8; ++i) { ts += rs[i]; tss += rss[i]; }

    const float invD = 1.0f / (float)D;
    const float mu   = ts * invD;
    const float var  = fmaxf(tss * invD - mu * mu, 0.0f);
    const float inv  = rsqrtf(var + LN_EPS);

    float4 gv = *reinterpret_cast<const float4*>(g + c);
    float4 bv = *reinterpret_cast<const float4*>(be + c);
    v.x = (v.x - mu) * inv * gv.x + bv.x;
    v.y = (v.y - mu) * inv * gv.y + bv.y;
    v.z = (v.z - mu) * inv * gv.z + bv.z;
    v.w = (v.w - mu) * inv * gv.w + bv.w;
    *reinterpret_cast<float4*>(zr + c) = v;
}

// ---------------------------------------------------------------------------
extern "C" void kernel_launch(void* const* d_in, const int* in_sizes, int n_in,
                              void* d_out, int out_size, void* d_ws, size_t ws_size,
                              hipStream_t stream)
{
    const float* hidden = (const float*)d_in[0];
    const float* input  = (const float*)d_in[1];
    const float* W      = (const float*)d_in[2];
    const float* b      = (const float*)d_in[3];
    const float* ln_w   = (const float*)d_in[4];
    const float* ln_b   = (const float*)d_in[5];

    const int       D = in_sizes[3];                 // 1024
    const long long M = (long long)in_sizes[0] / D;  // 16384

    // Workspace layout
    char*      ws    = (char*)d_ws;
    _Float16*  Xh    = (_Float16*)ws;                               // M*D f16
    _Float16*  Wq    = (_Float16*)(ws + (size_t)M * D * 2);         // D*D f16
    float*     scale = (float*)   (ws + (size_t)M * D * 2
                                      + (size_t)D * D * 2);         // D f32

    // 1) convert activations to f16 (L2-resident: 32 MB)
    const long long nElem = M * (long long)D;
    cvt_f32_to_f16_kernel<<<(unsigned)(nElem / 1024), 256, 0, stream>>>(
        hidden, Xh, nElem);

    // 2) quantize weights (integer-valued f16 + per-row scale)
    quant_weight_kernel<<<D, 256, 0, stream>>>(W, Wq, scale, D);

    // 3) WMMA GEMM + fused scale/bias/residual -> d_out holds z
    dim3 grid((unsigned)(D / 64), (unsigned)(M / 128));
    gemm_wmma_kernel<<<grid, 256, 0, stream>>>(
        Xh, Wq, scale, b, input, (float*)d_out, (int)M, D, D);

    // 4) in-place LayerNorm over last dim
    layernorm_kernel<<<(unsigned)M, D / 4, 0, stream>>>(
        (float*)d_out, ln_w, ln_b, D);
}
